// LigerLMHeadMSE_28681791602729
// MI455X (gfx1250) — matrix operand order repaired and checked by reading, not compiled, not run on hardware
//
#include <hip/hip_runtime.h>
#include <hip/hip_bf16.h>

typedef __attribute__((ext_vector_type(16))) __bf16 v16bf;
typedef __attribute__((ext_vector_type(8)))  float  v8f;

#define N_DIM 2048
#define H_DIM 1024
#define V_DIM 32000
#define BM 128
#define BN 128
#define BK 32
#define KT_STEPS (H_DIM / BK)          // 32
#define TILES_M (N_DIM / BM)           // 16
#define TILES_N (V_DIM / BN)           // 250
#define NPARTIALS (TILES_M * TILES_N)  // 4000

// Pack two f32 -> two bf16 (round-to-nearest-even) in one u32.
__device__ __forceinline__ unsigned pack_bf16x2(float a, float b) {
    unsigned ua = __float_as_uint(a);
    unsigned ub = __float_as_uint(b);
    ua += 0x7fffu + ((ua >> 16) & 1u);
    ub += 0x7fffu + ((ub >> 16) & 1u);
    return (ua >> 16) | (ub & 0xffff0000u);
}

// CDNA5 async copy: 16 bytes global -> LDS, tracked by ASYNCcnt (no VGPR data).
__device__ __forceinline__ void async_copy16(unsigned lds_off, const void* gaddr) {
    asm volatile("global_load_async_to_lds_b128 %0, %1, off"
                 :: "v"(lds_off), "v"((unsigned long long)(uintptr_t)gaddr)
                 : "memory");
}
__device__ __forceinline__ void wait_async0() {
    asm volatile("s_wait_asynccnt 0x0" ::: "memory");
}
__device__ __forceinline__ unsigned lds_off32(const void* p) {
    // Generic LDS pointer = {SHARED_BASE aperture, offset[31:0]} -> low 32 bits.
    return (unsigned)(uintptr_t)p;
}

// ---------------------------------------------------------------------------
// Pre-pass: convert f32 -> bf16 and scatter into WMMA *fragment order*:
//   dst[((row/16)*KT_STEPS + kt)*256 + lane*8 + p]   (1KB per 16x32 fragment)
// IS_A uses the 16-bit A-matrix 16x32 layout; !IS_A the B-matrix 32x16 layout.
// ---------------------------------------------------------------------------
template <bool IS_A>
__global__ __launch_bounds__(256)
void pack_frag_kernel(const float* __restrict__ src, unsigned* __restrict__ dst,
                      int npairs) {
    for (int q = blockIdx.x * blockDim.x + threadIdx.x; q < npairs;
         q += gridDim.x * blockDim.x) {
        const int k2g = q & 511;          // bf16-pair index within row (H/2)
        const int row = q >> 9;
        const int kt  = k2g >> 4;         // which K-block of 32
        const int k2  = k2g & 15;         // pair within K-block
        const float2 v = *(const float2*)(src + ((size_t)row << 10) + (k2g << 1));
        const unsigned pk = pack_bf16x2(v.x, v.y);
        int lane, p;
        if (IS_A) {
            lane = ((k2 & 4) ? 16 : 0) + (row & 15);
            p    = ((k2 >= 8) ? 4 : 0) + (k2 & 3);
        } else {
            lane = ((k2 & 4) ? 16 : 0) + (row & 15);
            p    = ((k2 >= 8) ? 4 : 0) + (k2 & 3);
        }
        dst[((((size_t)(row >> 4)) * KT_STEPS + kt) << 8) + lane * 8 + p] = pk;
    }
}

// ---------------------------------------------------------------------------
// Packed-operand GEMM + fused MSE, async double-buffered LDS pipeline.
// Block = 256 threads (8 waves). Tile = 128x128, K-step 32.
// ---------------------------------------------------------------------------
__global__ __launch_bounds__(256)
void lmhead_gemm_packed_kernel(const unsigned* __restrict__ xp,
                               const unsigned* __restrict__ wp,
                               const float* __restrict__ y,
                               float* __restrict__ partials) {
    __shared__ __align__(16) unsigned ldsA[2][2048];   // 2 x 8KB
    __shared__ __align__(16) unsigned ldsB[2][2048];   // 2 x 8KB
    __shared__ float red[256];

    const int tid  = threadIdx.x;
    const int lane = tid & 31;
    const int w    = tid >> 5;
    const int wm   = w & 3;     // 4 waves along M
    const int wn   = w >> 2;    // 2 waves along N

    const int bn0 = blockIdx.x * BN;
    const int bm0 = blockIdx.y * BM;
    const int mf0 = bm0 >> 4;   // first 16-row fragment index
    const int nf0 = bn0 >> 4;

    v8f acc[2][4] = {};

    // Issue async copies of one K-block tile pair into buffer `buf`.
    auto issue = [&](int buf, int kt) {
        #pragma unroll
        for (int c = 0; c < 2; ++c) {
            const int ch = tid + c * 256;   // 0..511 : 16B chunks of 8KB tile
            const int f  = ch >> 6;         // fragment 0..7
            const int r  = ch & 63;         // 16B chunk within 1KB fragment
            const unsigned* ga =
                xp + ((((size_t)(mf0 + f)) * KT_STEPS + kt) << 8) + r * 4;
            async_copy16(lds_off32(&ldsA[buf][f * 256 + r * 4]), ga);
            const unsigned* gb =
                wp + ((((size_t)(nf0 + f)) * KT_STEPS + kt) << 8) + r * 4;
            async_copy16(lds_off32(&ldsB[buf][f * 256 + r * 4]), gb);
        }
    };

    issue(0, 0);

    #pragma unroll 2
    for (int kt = 0; kt < KT_STEPS; ++kt) {
        const int cur = kt & 1;
        wait_async0();        // my copies into buf[cur] complete
        __syncthreads();      // everyone's copies complete; prev compute done
        if (kt + 1 < KT_STEPS) issue(cur ^ 1, kt + 1);

        v16bf afrag[2];
        v16bf bfrag[4];
        #pragma unroll
        for (int i = 0; i < 2; ++i)
            afrag[i] = *(const v16bf*)(&ldsA[cur][(2 * wm + i) * 256 + lane * 8]);
        #pragma unroll
        for (int j = 0; j < 4; ++j)
            bfrag[j] = *(const v16bf*)(&ldsB[cur][(4 * wn + j) * 256 + lane * 8]);

        #pragma unroll
        for (int i = 0; i < 2; ++i)
            #pragma unroll
            for (int j = 0; j < 4; ++j)
                acc[i][j] = __builtin_amdgcn_wmma_f32_16x16x32_bf16(
                    false, afrag[i], false, bfrag[j],
                    (short)0, acc[i][j], false, false);
    }

    // Epilogue: stream y non-temporally (keep W resident in L2), sum diff^2.
    const int n_in  = lane & 15;
    const int m_off = (lane >> 4) * 8;
    float local = 0.0f;
    #pragma unroll
    for (int i = 0; i < 2; ++i) {
        #pragma unroll
        for (int j = 0; j < 4; ++j) {
            const int gm_base = bm0 + (2 * wm + i) * 16 + m_off;
            const int gn      = bn0 + (4 * wn + j) * 16 + n_in;
            #pragma unroll
            for (int r = 0; r < 8; ++r) {
                const float yy =
                    __builtin_nontemporal_load(y + (size_t)(gm_base + r) * V_DIM + gn);
                const float d = acc[i][j][r] - yy;
                local = fmaf(d, d, local);
            }
        }
    }

    red[tid] = local;
    __syncthreads();
    #pragma unroll
    for (int off = 128; off > 0; off >>= 1) {
        if (tid < off) red[tid] += red[tid + off];
        __syncthreads();
    }
    if (tid == 0)
        partials[blockIdx.y * gridDim.x + blockIdx.x] = red[0];
}

// ---------------------------------------------------------------------------
// Fallback fused kernel (round-1 version): on-the-fly conversion, no workspace
// beyond the partial array. Used only if ws_size is too small for packing.
// ---------------------------------------------------------------------------
__global__ __launch_bounds__(256)
void lmhead_mse_gemm_kernel(const float* __restrict__ x,
                            const float* __restrict__ y,
                            const float* __restrict__ W,
                            float* __restrict__ partials) {
    __shared__ __align__(16) unsigned ldsA[8 * 256];
    __shared__ __align__(16) unsigned ldsB[8 * 256];
    __shared__ float red[256];

    const int tid  = threadIdx.x;
    const int lane = tid & 31;
    const int w    = tid >> 5;
    const int wm   = w & 3;
    const int wn   = w >> 2;
    const int bn0 = blockIdx.x * BN;
    const int bm0 = blockIdx.y * BM;

    v8f acc[2][4] = {};

    for (int kt = 0; kt < KT_STEPS; ++kt) {
        const int k0 = kt * BK;
        #pragma unroll
        for (int it = 0; it < 8; ++it) {
            const int q  = tid + it * 256;
            const int k2 = q & 15;
            const int m  = q >> 4;
            const float2 v =
                *(const float2*)(x + (size_t)(bm0 + m) * H_DIM + k0 + 2 * k2);
            const unsigned pk = pack_bf16x2(v.x, v.y);
            const int alane = ((k2 & 4) ? 16 : 0) + (m & 15);
            const int p = ((k2 >= 8) ? 4 : 0) + (k2 & 3);
            ldsA[(m >> 4) * 256 + alane * 8 + p] = pk;
        }
        #pragma unroll
        for (int it = 0; it < 8; ++it) {
            const int q  = tid + it * 256;
            const int k2 = q & 15;
            const int n  = q >> 4;
            const float2 v =
                *(const float2*)(W + (size_t)(bn0 + n) * H_DIM + k0 + 2 * k2);
            const unsigned pk = pack_bf16x2(v.x, v.y);
            const int blane = ((k2 & 4) ? 16 : 0) + (n & 15);
            const int p = ((k2 >= 8) ? 4 : 0) + (k2 & 3);
            ldsB[(n >> 4) * 256 + blane * 8 + p] = pk;
        }
        __syncthreads();

        v16bf afrag[2];
        v16bf bfrag[4];
        #pragma unroll
        for (int i = 0; i < 2; ++i)
            afrag[i] = *(const v16bf*)(&ldsA[(2 * wm + i) * 256 + lane * 8]);
        #pragma unroll
        for (int j = 0; j < 4; ++j)
            bfrag[j] = *(const v16bf*)(&ldsB[(4 * wn + j) * 256 + lane * 8]);
        #pragma unroll
        for (int i = 0; i < 2; ++i)
            #pragma unroll
            for (int j = 0; j < 4; ++j)
                acc[i][j] = __builtin_amdgcn_wmma_f32_16x16x32_bf16(
                    false, afrag[i], false, bfrag[j],
                    (short)0, acc[i][j], false, false);
        __syncthreads();
    }

    const int n_in  = lane & 15;
    const int m_off = (lane >> 4) * 8;
    float local = 0.0f;
    #pragma unroll
    for (int i = 0; i < 2; ++i)
        #pragma unroll
        for (int j = 0; j < 4; ++j) {
            const int gm_base = bm0 + (2 * wm + i) * 16 + m_off;
            const int gn      = bn0 + (4 * wn + j) * 16 + n_in;
            #pragma unroll
            for (int r = 0; r < 8; ++r) {
                const float yy =
                    __builtin_nontemporal_load(y + (size_t)(gm_base + r) * V_DIM + gn);
                const float d = acc[i][j][r] - yy;
                local = fmaf(d, d, local);
            }
        }

    red[tid] = local;
    __syncthreads();
    #pragma unroll
    for (int off = 128; off > 0; off >>= 1) {
        if (tid < off) red[tid] += red[tid + off];
        __syncthreads();
    }
    if (tid == 0)
        partials[blockIdx.y * gridDim.x + blockIdx.x] = red[0];
}

// ---------------------------------------------------------------------------
// Final deterministic reduction.
// ---------------------------------------------------------------------------
__global__ __launch_bounds__(256)
void lmhead_mse_reduce_kernel(const float* __restrict__ partials,
                              float* __restrict__ out) {
    __shared__ float red[256];
    float s = 0.0f;
    for (int i = threadIdx.x; i < NPARTIALS; i += 256) s += partials[i];
    red[threadIdx.x] = s;
    __syncthreads();
    #pragma unroll
    for (int off = 128; off > 0; off >>= 1) {
        if (threadIdx.x < off) red[threadIdx.x] += red[threadIdx.x + off];
        __syncthreads();
    }
    if (threadIdx.x == 0)
        out[0] = red[0] * (1.0f / ((float)N_DIM * (float)V_DIM));
}

extern "C" void kernel_launch(void* const* d_in, const int* in_sizes, int n_in,
                              void* d_out, int out_size, void* d_ws, size_t ws_size,
                              hipStream_t stream) {
    const float* x = (const float*)d_in[0];   // [2048, 1024]
    const float* y = (const float*)d_in[1];   // [2048, 32000]
    const float* W = (const float*)d_in[2];   // [32000, 1024]
    float* out = (float*)d_out;

    const size_t xp_u32 = (size_t)N_DIM * H_DIM / 2;   // 1,048,576 u32 (4 MB)
    const size_t wp_u32 = (size_t)V_DIM * H_DIM / 2;   // 16,384,000 u32 (65.5 MB)
    const size_t need   = (xp_u32 + wp_u32) * 4 + NPARTIALS * 4;

    dim3 grid(TILES_N, TILES_M);
    dim3 block(256);

    if (ws_size >= need) {
        unsigned* xp = (unsigned*)d_ws;
        unsigned* wp = xp + xp_u32;
        float* partials = (float*)(wp + wp_u32);

        pack_frag_kernel<true><<<4096, 256, 0, stream>>>(x, xp, (int)xp_u32);
        pack_frag_kernel<false><<<8192, 256, 0, stream>>>(W, wp, (int)wp_u32);
        lmhead_gemm_packed_kernel<<<grid, block, 0, stream>>>(xp, wp, y, partials);
        lmhead_mse_reduce_kernel<<<1, 256, 0, stream>>>(partials, out);
    } else {
        float* partials = (float*)d_ws;  // 16 KB
        lmhead_mse_gemm_kernel<<<grid, block, 0, stream>>>(x, y, W, partials);
        lmhead_mse_reduce_kernel<<<1, 256, 0, stream>>>(partials, out);
    }
}
